// MultiHeadSelfAttention_13718125543701
// MI455X (gfx1250) — matrix operand order repaired and checked by reading, not compiled, not run on hardware
//
#include <hip/hip_runtime.h>
#include <hip/hip_bf16.h>

// MI455X / gfx1250, wave32. WMMA f32_16x16x32_f16 pipeline:
//   K1  prep_xt   : xt = x + pos_enc_2d  -> f16 (b, n=4096, c=64)
//   K1b prep_w    : Wq/Wk/Wv f32 -> f16, transposed (d, c) for row-contig B loads
//   K2  proj      : Q(=/8), K row-major f16 ; V stored transposed (d, n) f16
//   K3  colstats  : per-column (softmax over dim=1!) online max & sum of exp.
//                   4 waves/block share a double-buffered LDS Q stream; 256 blocks
//                   so the grid covers all WGPs (occupancy > sharing factor here).
//   K4  attn_out  : recompute S tiles from LDS-staged K chunks, P = exp(S-cm)/cs
//                   (f16 via per-wave LDS transpose + s_wait_dscnt), O += P @ V with
//                   V chunks LDS-staged; software-pipelined global->LDS + prefetch.
//
// Workspace layout (needs ~8.2 MB):
//   [0,2MB)   XT f16   [2,4MB) Qs f16   [4,6MB) Km f16   [6,8MB) Vt f16
//   [8MB..)   Wt f16 (3*4096), colmax f32 (16K), colsum f32 (16K)

typedef __attribute__((ext_vector_type(16))) _Float16 v16h;
typedef __attribute__((ext_vector_type(8)))  float    v8f;
typedef __attribute__((ext_vector_type(4)))  float    f32x4;

union V16H { v16h v; f32x4 f4[2]; };

// A-matrix 16x32 f16 tile (ISA 7.12.2): lane<16 -> row M=lane, K in {kb..kb+7, 16+kb..16+kb+7}
// with kb = 8*(lane>=16). Source is row-major with leading dim ld; base already at (row0, k0).
__device__ __forceinline__ v16h load_tileA(const _Float16* base, int ld, int lane) {
    const int m  = lane & 15;
    const int kb = (lane >> 4) << 3;          // 0 or 8
    const _Float16* r = base + (size_t)m * ld;
    V16H u;
    u.f4[0] = *(const f32x4*)(r + kb);        // K = kb..kb+7
    u.f4[1] = *(const f32x4*)(r + 16 + kb);   // K = 16+kb..16+kb+7
    return u.v;
}

// B-matrix 32x16 f16 tile: lane<16 -> column N=lane, K in {kb..kb+7, 16+kb..16+kb+7}, kb = 8*(lane>=16).
// Source holds B transposed (i.e. Src[col][k] row-major, ld = stride of col rows),
// so each lane does two contiguous 16B loads.
__device__ __forceinline__ v16h load_tileB_T(const _Float16* base, int ld, int lane) {
    const int col = lane & 15;
    const int k0  = (lane >> 4) << 3;         // 0 or 8
    const _Float16* r = base + (size_t)col * ld;
    V16H u;
    u.f4[0] = *(const f32x4*)(r + k0);        // K = k0..k0+7
    u.f4[1] = *(const f32x4*)(r + 16 + k0);   // K = 16+k0..16+k0+7
    return u.v;
}

__device__ __forceinline__ v8f wmma16x16x32(v16h a, v16h b, v8f c) {
    return __builtin_amdgcn_wmma_f32_16x16x32_f16(false, a, false, b, (short)0, c, false, false);
}

#define VZERO8 {0.f,0.f,0.f,0.f,0.f,0.f,0.f,0.f}

// ---------------- K1: positional encoding + transpose to (b, n, c) f16 ----------------
__global__ void prep_xt(const float* __restrict__ x, _Float16* __restrict__ xt) {
    const int idx = blockIdx.x * blockDim.x + threadIdx.x;   // b*4096*64 = 1M
    const int cch = idx & 63;
    const int n   = (idx >> 6) & 4095;
    const int bi  = idx >> 18;
    const int y = n >> 6, xc = n & 63;
    const float K = 9.210340371976184f / 16.f;               // ln(10000)/16
    float pe;
    if      (cch < 16) pe = sinf((float)y  * expf(-(float)(cch     ) * K));
    else if (cch < 32) pe = cosf((float)y  * expf(-(float)(cch - 16) * K));
    else if (cch < 48) pe = sinf((float)xc * expf(-(float)(cch - 32) * K));
    else               pe = cosf((float)xc * expf(-(float)(cch - 48) * K));
    const float v = x[(((size_t)bi * 64 + cch) * 64 + y) * 64 + xc] + pe;
    xt[((size_t)bi * 4096 + n) * 64 + cch] = (_Float16)v;
}

// ---------------- K1b: weights f32 -> f16, transposed to (d, c) ----------------
__global__ void prep_w(const float* __restrict__ wq, const float* __restrict__ wk,
                       const float* __restrict__ wv, _Float16* __restrict__ wt) {
    const int idx = blockIdx.x * blockDim.x + threadIdx.x;   // 3*4096
    const int which = idx >> 12;
    const int r = idx & 4095;
    const int k = r >> 6, d = r & 63;
    const float* src = (which == 0) ? wq : (which == 1) ? wk : wv;
    wt[which * 4096 + d * 64 + k] = (_Float16)src[k * 64 + d];
}

// ---------------- K2: Q,K,V projections. One wave per 16x16 output tile ----------------
__global__ void proj(const _Float16* __restrict__ xt, const _Float16* __restrict__ wt,
                     _Float16* __restrict__ qs, _Float16* __restrict__ km,
                     _Float16* __restrict__ vt) {
    const int lane = threadIdx.x & 31;
    int wid = blockIdx.x * (blockDim.x >> 5) + (threadIdx.x >> 5);  // 12288 waves
    const int which = wid % 3;  wid /= 3;
    const int dt = wid & 3;     wid >>= 2;
    const int nt = wid & 255;
    const int bi = wid >> 8;

    const _Float16* A = xt + ((size_t)bi * 4096 + nt * 16) * 64;
    const _Float16* B = wt + which * 4096 + dt * 16 * 64;           // Wt rows = d
    v16h a0 = load_tileA(A,      64, lane);
    v16h a1 = load_tileA(A + 32, 64, lane);
    v16h b0 = load_tileB_T(B,      64, lane);
    v16h b1 = load_tileB_T(B + 32, 64, lane);
    v8f c = VZERO8;
    c = wmma16x16x32(a0, b0, c);
    c = wmma16x16x32(a1, b1, c);

    const int ncol = lane & 15;
    const int r0   = (lane >> 4) << 3;
    if (which == 0) {       // Q, pre-scaled by 1/sqrt(64)
        _Float16* dst = qs + ((size_t)bi * 4096 + nt * 16) * 64 + dt * 16;
        #pragma unroll
        for (int r = 0; r < 8; ++r) dst[(size_t)(r0 + r) * 64 + ncol] = (_Float16)(c[r] * 0.125f);
    } else if (which == 1) { // K row-major
        _Float16* dst = km + ((size_t)bi * 4096 + nt * 16) * 64 + dt * 16;
        #pragma unroll
        for (int r = 0; r < 8; ++r) dst[(size_t)(r0 + r) * 64 + ncol] = (_Float16)c[r];
    } else {                 // V transposed: Vt[b][d][n]
        _Float16* dst = vt + (size_t)bi * 64 * 4096;
        #pragma unroll
        for (int r = 0; r < 8; ++r)
            dst[(size_t)(dt * 16 + ncol) * 4096 + nt * 16 + r0 + r] = (_Float16)c[r];
    }
}

// ---------------- K3: per-column softmax stats (softmax over QUERY axis) ----------------
// 4 waves/block own 4 adjacent 16-column groups of the SAME batch and share the Q stream:
// 32-row Q chunks staged through double-buffered LDS with a register pipeline.
// 256 blocks (blockDim=128) to cover all WGPs; 8KB LDS per block.
__global__ void colstats(const _Float16* __restrict__ qs, const _Float16* __restrict__ km,
                         float* __restrict__ cmax, float* __restrict__ csum) {
    __shared__ _Float16 lds_q[2][32][64];                    // 2 x 4KB
    const int t    = threadIdx.x;                            // 0..127
    const int lane = t & 31;
    const int wv   = t >> 5;                                 // 0..3
    const int bi   = blockIdx.x >> 6;                        // 64 blocks per batch
    const int mt   = ((blockIdx.x & 63) << 2) + wv;

    const _Float16* Bb = km + ((size_t)bi * 4096 + mt * 16) * 64;   // B = K^T: rows of K
    v16h b0 = load_tileB_T(Bb,      64, lane);
    v16h b1 = load_tileB_T(Bb + 32, 64, lane);
    const _Float16* Qb = qs + (size_t)bi * 4096 * 64;

    // cooperative 32x64 chunk: 128 threads x 2 x 16B. slot t -> row t>>3, col (t&7)*8
    const int qr = t >> 3, qc = (t & 7) * 8;
    f32x4 rQ0 = *(const f32x4*)(Qb + (size_t)qr * 64 + qc);
    f32x4 rQ1 = *(const f32x4*)(Qb + (size_t)(16 + qr) * 64 + qc);

    float runmax = -3.4e38f, runsum = 0.f;
    for (int it = 0; it < 128; ++it) {
        const int sB = it & 1;
        *(f32x4*)&lds_q[sB][qr][qc]      = rQ0;
        *(f32x4*)&lds_q[sB][16 + qr][qc] = rQ1;
        __syncthreads();
        if (it < 127) {
            rQ0 = *(const f32x4*)(Qb + (size_t)((it + 1) * 32 + qr) * 64 + qc);
            rQ1 = *(const f32x4*)(Qb + (size_t)((it + 1) * 32 + 16 + qr) * 64 + qc);
        }
        if (it < 126)
            __builtin_prefetch(Qb + (size_t)((it + 2) * 32 + qr) * 64 + qc, 0, 1);
        const _Float16* lq = &lds_q[sB][0][0];
        #pragma unroll
        for (int half = 0; half < 2; ++half) {
            const _Float16* A = lq + half * 16 * 64;
            v16h a0 = load_tileA(A,      64, lane);
            v16h a1 = load_tileA(A + 32, 64, lane);
            v8f s = VZERO8;
            s = wmma16x16x32(a0, b0, s);
            s = wmma16x16x32(a1, b1, s);
            float tm = s[0];
            #pragma unroll
            for (int r = 1; r < 8; ++r) tm = fmaxf(tm, s[r]);
            tm = fmaxf(tm, __shfl_xor(tm, 16, 32));          // combine row-halves (same col)
            const float nm = fmaxf(runmax, tm);
            float partial = 0.f;
            #pragma unroll
            for (int r = 0; r < 8; ++r) partial += __expf(s[r] - nm);
            partial += __shfl_xor(partial, 16, 32);
            runsum = runsum * __expf(runmax - nm) + partial;
            runmax = nm;
        }
        __syncthreads();
    }
    if (lane < 16) {
        cmax[(size_t)bi * 4096 + mt * 16 + lane] = runmax;
        csum[(size_t)bi * 4096 + mt * 16 + lane] = runsum;
    }
}

// ---------------- K4: O = softmax(S) @ V, recomputing S tiles ----------------
// 4 waves/block own 4 adjacent 16-row output groups of the SAME batch; K and Vt chunks
// staged through double-buffered LDS (shared by the 4 waves), register-pipelined.
__global__ void attn_out(const _Float16* __restrict__ qs, const _Float16* __restrict__ km,
                         const _Float16* __restrict__ vt, const float* __restrict__ cmax,
                         const float* __restrict__ csum, float* __restrict__ out) {
    __shared__ _Float16 lds_k[2][32][64];                    // 2 x 4KB
    __shared__ _Float16 lds_v[2][64][32];                    // 2 x 4KB
    __shared__ _Float16 plds[4][16][32];                     // per-wave P tile, 4KB
    const int t    = threadIdx.x;                            // 0..127
    const int lane = t & 31;
    const int wv   = t >> 5;                                 // 0..3
    const int bi   = blockIdx.x >> 6;                        // 64 blocks per batch
    const int nt   = ((blockIdx.x & 63) << 2) + wv;
    const int n0   = nt * 16;

    const _Float16* A = qs + ((size_t)bi * 4096 + n0) * 64;
    v16h a0 = load_tileA(A,      64, lane);
    v16h a1 = load_tileA(A + 32, 64, lane);
    const _Float16* Kb  = km   + (size_t)bi * 4096 * 64;
    const _Float16* Vb  = vt   + (size_t)bi * 64 * 4096;
    const float*    cmb = cmax + (size_t)bi * 4096;
    const float*    csb = csum + (size_t)bi * 4096;

    const int ncol = lane & 15;
    const int r0   = (lane >> 4) << 3;
    v8f o0 = VZERO8, o1 = VZERO8, o2 = VZERO8, o3 = VZERO8;
    _Float16* myp = &plds[wv][0][0];

    // K chunk 32x64 and V chunk 64x32: 128 threads x 2 x 16B each
    const int kr = t >> 3, kc = (t & 7) * 8;
    const int vr = t >> 2, vc = (t & 3) * 8;
    f32x4 rK0 = *(const f32x4*)(Kb + (size_t)kr * 64 + kc);
    f32x4 rK1 = *(const f32x4*)(Kb + (size_t)(16 + kr) * 64 + kc);
    f32x4 rV0 = *(const f32x4*)(Vb + (size_t)vr * 4096 + vc);
    f32x4 rV1 = *(const f32x4*)(Vb + (size_t)(32 + vr) * 4096 + vc);

    for (int it = 0; it < 128; ++it) {
        const int sB = it & 1;
        const int m0 = it * 32;
        *(f32x4*)&lds_k[sB][kr][kc]      = rK0;
        *(f32x4*)&lds_k[sB][16 + kr][kc] = rK1;
        *(f32x4*)&lds_v[sB][vr][vc]      = rV0;
        *(f32x4*)&lds_v[sB][32 + vr][vc] = rV1;
        __syncthreads();
        if (it < 127) {
            rK0 = *(const f32x4*)(Kb + (size_t)((it + 1) * 32 + kr) * 64 + kc);
            rK1 = *(const f32x4*)(Kb + (size_t)((it + 1) * 32 + 16 + kr) * 64 + kc);
            rV0 = *(const f32x4*)(Vb + (size_t)vr * 4096 + (it + 1) * 32 + vc);
            rV1 = *(const f32x4*)(Vb + (size_t)(32 + vr) * 4096 + (it + 1) * 32 + vc);
        }
        if (it < 126) {
            __builtin_prefetch(Kb + (size_t)((it + 2) * 32 + kr) * 64 + kc, 0, 1);
            __builtin_prefetch(Vb + (size_t)vr * 4096 + (it + 2) * 32 + vc, 0, 1);
        }

        // S tile 16(n) x 32(m) as two 16x16 WMMA results, B tiles from LDS K chunk
        const _Float16* lk = &lds_k[sB][0][0];
        const _Float16* lv = &lds_v[sB][0][0];
        v16h k0a = load_tileB_T(lk,                64, lane);
        v16h k0b = load_tileB_T(lk + 32,           64, lane);
        v16h k1a = load_tileB_T(lk + 16 * 64,      64, lane);
        v16h k1b = load_tileB_T(lk + 16 * 64 + 32, 64, lane);
        v8f s0 = VZERO8, s1 = VZERO8;
        s0 = wmma16x16x32(a0, k0a, s0);
        s0 = wmma16x16x32(a1, k0b, s0);
        s1 = wmma16x16x32(a0, k1a, s1);
        s1 = wmma16x16x32(a1, k1b, s1);

        // column-normalized probabilities (columns m are per-lane in D layout)
        const float cm0 = cmb[m0 + ncol],      iv0 = 1.f / csb[m0 + ncol];
        const float cm1 = cmb[m0 + 16 + ncol], iv1 = 1.f / csb[m0 + 16 + ncol];
        #pragma unroll
        for (int r = 0; r < 8; ++r)
            myp[(size_t)(r0 + r) * 32 + ncol]      = (_Float16)(__expf(s0[r] - cm0) * iv0);
        #pragma unroll
        for (int r = 0; r < 8; ++r)
            myp[(size_t)(r0 + r) * 32 + 16 + ncol] = (_Float16)(__expf(s1[r] - cm1) * iv1);

        asm volatile("s_wait_dscnt 0" ::: "memory");  // D-layout stores -> A-layout reads
        v16h pa = load_tileA(myp, 32, lane);          // P as next A matrix (K = 32 m's)

        o0 = wmma16x16x32(pa, load_tileB_T(lv,           32, lane), o0);
        o1 = wmma16x16x32(pa, load_tileB_T(lv + 16 * 32, 32, lane), o1);
        o2 = wmma16x16x32(pa, load_tileB_T(lv + 32 * 32, 32, lane), o2);
        o3 = wmma16x16x32(pa, load_tileB_T(lv + 48 * 32, 32, lane), o3);
        __syncthreads();
    }

    // out (b, c, h*w): out[bi][d][n] = O[n][d]
    float* ob = out + (size_t)bi * 64 * 4096;
    #pragma unroll
    for (int r = 0; r < 8; ++r) {
        ob[(size_t)( 0 + ncol) * 4096 + n0 + r0 + r] = o0[r];
        ob[(size_t)(16 + ncol) * 4096 + n0 + r0 + r] = o1[r];
        ob[(size_t)(32 + ncol) * 4096 + n0 + r0 + r] = o2[r];
        ob[(size_t)(48 + ncol) * 4096 + n0 + r0 + r] = o3[r];
    }
}

extern "C" void kernel_launch(void* const* d_in, const int* in_sizes, int n_in,
                              void* d_out, int out_size, void* d_ws, size_t ws_size,
                              hipStream_t stream) {
    (void)in_sizes; (void)n_in; (void)out_size; (void)ws_size;
    const float* x  = (const float*)d_in[0];
    const float* Wq = (const float*)d_in[1];
    const float* Wk = (const float*)d_in[2];
    const float* Wv = (const float*)d_in[3];

    char* ws = (char*)d_ws;
    _Float16* XT = (_Float16*)(ws);
    _Float16* Qs = (_Float16*)(ws + (size_t)(2u << 20));
    _Float16* Km = (_Float16*)(ws + (size_t)(4u << 20));
    _Float16* Vt = (_Float16*)(ws + (size_t)(6u << 20));
    _Float16* Wt = (_Float16*)(ws + (size_t)(8u << 20));               // 24 KB
    float* cmax  = (float*)  (ws + (size_t)(8u << 20) + (64u << 10));  // 64 KB
    float* csum  = (float*)  (ws + (size_t)(8u << 20) + (128u << 10)); // 64 KB

    prep_xt <<<4096, 256, 0, stream>>>(x, XT);
    prep_w  <<<48,   256, 0, stream>>>(Wq, Wk, Wv, Wt);
    proj    <<<1536, 256, 0, stream>>>(XT, Wt, Qs, Km, Vt);
    colstats<<<256,  128, 0, stream>>>(Qs, Km, cmax, csum);
    attn_out<<<256,  128, 0, stream>>>(Qs, Km, Vt, cmax, csum, (float*)d_out);
}
